// GatedDeltaNetAttention_29643864277630
// MI455X (gfx1250) — compile-verified, matrix operand order checked by reading, not run on hardware
//
#include <hip/hip_runtime.h>
#include <math.h>

// ---------------------------------------------------------------------------
// GatedDeltaNet on MI455X (gfx1250, wave32).
// GEMMs: v_wmma_f32_16x16x32_bf16, fed by double-buffered async global->LDS
// copies (ASYNCcnt) with ds_load_b128 fragment reads. Recurrent scan is
// register-resident (state row per lane). Workspace ~400MB (see alloc()).
// ---------------------------------------------------------------------------

typedef __attribute__((ext_vector_type(16))) __bf16         v16bf;
typedef __attribute__((ext_vector_type(8)))  float          v8f;
typedef __attribute__((ext_vector_type(8)))  unsigned short u16x8;
typedef __attribute__((ext_vector_type(4)))  int            i32x4;

union BFrag {
  v16bf v;
  u16x8 h[2];
};

#if defined(__has_builtin)
#if __has_builtin(__builtin_amdgcn_global_load_async_to_lds_b128) && \
    __has_builtin(__builtin_amdgcn_s_wait_asynccnt)
#define HAVE_ASYNC_LDS 1
#endif
#endif
#ifndef HAVE_ASYNC_LDS
#define HAVE_ASYNC_LDS 0
#endif

#if HAVE_ASYNC_LDS
typedef __attribute__((address_space(1))) i32x4 gvec_t;   // global int4
typedef __attribute__((address_space(3))) i32x4 lvec_t;   // LDS int4
#endif

__device__ __forceinline__ unsigned short f2bf(float f) {
  unsigned int u = __float_as_uint(f);
  u += 0x7FFFu + ((u >> 16) & 1u);   // round-to-nearest-even
  return (unsigned short)(u >> 16);
}
__device__ __forceinline__ float bf2f(unsigned short h) {
  return __uint_as_float(((unsigned int)h) << 16);
}
__device__ __forceinline__ float sigm(float x) { return 1.f / (1.f + __expf(-x)); }

// ---------------------------------------------------------------- converters
__global__ __launch_bounds__(256) void k_f2bf(const float* __restrict__ s,
                                              unsigned short* __restrict__ d, int n) {
  int i = blockIdx.x * 256 + threadIdx.x;
  if (i < n) d[i] = f2bf(s[i]);
}

// src: K x N row-major fp32  ->  dst: N x K row-major bf16
__global__ __launch_bounds__(256) void k_transpose_bf(const float* __restrict__ src,
                                                      unsigned short* __restrict__ dst,
                                                      int K, int N) {
  int i = blockIdx.x * 256 + threadIdx.x;
  if (i >= K * N) return;
  int k = i / N, n = i % N;
  dst[(size_t)n * K + k] = f2bf(src[i]);
}

__global__ void k_concat_bias(const float* __restrict__ a, const float* __restrict__ b,
                              float* __restrict__ d) {
  int t = threadIdx.x;
  if (t < 32) d[t] = (t < 16) ? a[t] : b[t - 16];
}

// --------------------------------------------------------------- WMMA GEMM
// C[M,N] = act( A[M,K](bf16) * Bt[N,K](bf16)^T + bias )  mode: 0=none 1=sigmoid 2=silu
// 256 thr = 8 waves; block tile 128x128, wave tile 64x32 (4x2 WMMA), BK=32,
// double-buffered async global->LDS staging. LDS rows padded to 40 ushorts.
#define LDSROW 40

__device__ __forceinline__ void stage16(const unsigned short* g, unsigned short* l) {
#if HAVE_ASYNC_LDS
  __builtin_amdgcn_global_load_async_to_lds_b128(
      (gvec_t*)(uintptr_t)g, (lvec_t*)(uintptr_t)l, 0, 0);
#else
  *(u16x8*)l = *(const u16x8*)g;
#endif
}

// stage a 128-row x 32-col bf16 tile (4 x 16B chunks per row), 2 chunks/thread
__device__ __forceinline__ void stage_tile(const unsigned short* __restrict__ gbase,
                                           int ldg, int rowClamp,
                                           unsigned short* lbase, int tid) {
#pragma unroll
  for (int it = 0; it < 2; ++it) {
    const int i   = tid + it * 256;
    int row       = i >> 2;
    const int kc  = i & 3;
    const int gr  = (row < rowClamp) ? row : (rowClamp - 1);
    stage16(gbase + (size_t)gr * ldg + kc * 8, lbase + row * LDSROW + kc * 8);
  }
}

__device__ __forceinline__ void wait_stage() {
#if HAVE_ASYNC_LDS
  __builtin_amdgcn_s_wait_asynccnt(0);
#endif
  __syncthreads();
}

__global__ __launch_bounds__(256) void k_gemm_bf16(const unsigned short* __restrict__ A,
                                                   const unsigned short* __restrict__ Bt,
                                                   const float* __restrict__ bias,
                                                   float* __restrict__ C,
                                                   int M, int N, int K, int mode) {
  __shared__ __align__(16) unsigned short As[2][128 * LDSROW];
  __shared__ __align__(16) unsigned short Bs[2][128 * LDSROW];

  const int tid  = threadIdx.x;
  const int lane = tid & 31;
  const int wid  = tid >> 5;
  const int l16  = lane & 15;
  const int hiK  = (lane >> 4) & 1;          // lanes 16-31 take K 8-15 / 24-31
  const int blockM = blockIdx.y * 128;
  const int blockN = blockIdx.x * 128;
  const int waveM  = (wid >> 2) * 64;        // 0 or 64
  const int waveN  = (wid & 3) * 32;         // 0,32,64,96

  const unsigned short* Abase = A  + (size_t)blockM * K;
  const unsigned short* Bbase = Bt + (size_t)blockN * K;
  int bRows = N - blockN;
  if (bRows > 128) bRows = 128;
  if (bRows < 1)   bRows = 1;

  v8f acc[4][2] = {};

  stage_tile(Abase, K, 128,   As[0], tid);
  stage_tile(Bbase, K, bRows, Bs[0], tid);
  wait_stage();

  int cur = 0;
  for (int k0 = 0; k0 < K; k0 += 32) {
    if (k0 + 32 < K) {                      // prefetch next K slice into other buffer
      stage_tile(Abase + k0 + 32, K, 128,   As[cur ^ 1], tid);
      stage_tile(Bbase + k0 + 32, K, bRows, Bs[cur ^ 1], tid);
    }
    BFrag af[4], bf[2];
#pragma unroll
    for (int i = 0; i < 4; ++i) {
      const u16x8* p = (const u16x8*)(As[cur] + (waveM + i * 16 + l16) * LDSROW + hiK * 8);
      af[i].h[0] = p[0];                    // K 0-7  (or 8-15)
      af[i].h[1] = p[2];                    // K 16-23 (or 24-31)
    }
#pragma unroll
    for (int j = 0; j < 2; ++j) {
      const u16x8* p = (const u16x8*)(Bs[cur] + (waveN + j * 16 + l16) * LDSROW + hiK * 8);
      bf[j].h[0] = p[0];
      bf[j].h[1] = p[2];
    }
#pragma unroll
    for (int i = 0; i < 4; ++i)
#pragma unroll
      for (int j = 0; j < 2; ++j)
        acc[i][j] = __builtin_amdgcn_wmma_f32_16x16x32_bf16(
            false, af[i].v, false, bf[j].v, (short)0, acc[i][j], false, false);
    wait_stage();
    cur ^= 1;
  }

#pragma unroll
  for (int j = 0; j < 2; ++j) {
    const int col = blockN + waveN + j * 16 + l16;
    if (col >= N) continue;
    const float bs = bias ? bias[col] : 0.f;
#pragma unroll
    for (int i = 0; i < 4; ++i) {
#pragma unroll
      for (int r = 0; r < 8; ++r) {
        const int row = blockM + waveM + i * 16 + ((lane >= 16) ? (r + 8) : r);
        float v = acc[i][j][r] + bs;
        if (mode == 1)      v = sigm(v);
        else if (mode == 2) v = v * sigm(v);
        C[(size_t)row * N + col] = v;
      }
    }
  }
}

// ----------------------------------------- causal conv(K=4) + SiLU + head L2
__global__ __launch_bounds__(256) void k_conv_act(const float* __restrict__ lin,
                                                  const float* __restrict__ w,
                                                  unsigned short* __restrict__ out,
                                                  int S, int do_norm) {
  const int row = blockIdx.x;            // b*S + s
  const int s   = row % S;
  const int tid = threadIdx.x;
  const int c0  = tid * 4;

  float acc[4] = {0.f, 0.f, 0.f, 0.f};
#pragma unroll
  for (int t = 0; t < 4; ++t) {
    const int ss = s - 3 + t;
    if (ss >= 0) {
      const float4 xv = *(const float4*)(lin + (size_t)(row - 3 + t) * 1024 + c0);
      acc[0] += xv.x * w[(c0 + 0) * 4 + t];
      acc[1] += xv.y * w[(c0 + 1) * 4 + t];
      acc[2] += xv.z * w[(c0 + 2) * 4 + t];
      acc[3] += xv.w * w[(c0 + 3) * 4 + t];
    }
  }
#pragma unroll
  for (int j = 0; j < 4; ++j) acc[j] = acc[j] * sigm(acc[j]);

  __shared__ float red[256];
  float scale = 1.f;
  if (do_norm) {                         // per-head (64ch = 16 threads) L2 norm
    red[tid] = acc[0]*acc[0] + acc[1]*acc[1] + acc[2]*acc[2] + acc[3]*acc[3];
    __syncthreads();
    if ((tid & 15) == 0) {
      float sm = 0.f;
      for (int j = 0; j < 16; ++j) sm += red[tid + j];
      red[tid] = sm;
    }
    __syncthreads();
    scale = 1.f / (sqrtf(red[tid & ~15]) + 1e-6f);
  }
  ushort4 pv;
  pv.x = f2bf(acc[0] * scale);
  pv.y = f2bf(acc[1] * scale);
  pv.z = f2bf(acc[2] * scale);
  pv.w = f2bf(acc[3] * scale);
  *(ushort4*)(out + (size_t)row * 1024 + c0) = pv;
}

// ------------------------------------------------------- gated delta-rule scan
__global__ __launch_bounds__(64) void k_scan(const unsigned short* __restrict__ qb,
                                             const unsigned short* __restrict__ kb,
                                             const unsigned short* __restrict__ vb,
                                             const float* __restrict__ ab,
                                             float* __restrict__ o, int S) {
  const int b = blockIdx.x >> 4;
  const int h = blockIdx.x & 15;
  const int d = threadIdx.x;

  float Srow[64];
#pragma unroll
  for (int e = 0; e < 64; ++e) Srow[e] = 0.f;

  __shared__ float lq[8][64], lk[8][64], lv[8][64];
  __shared__ float la[8], lbt[8];
  const size_t hoff = (size_t)h * 64;

  for (int s0 = 0; s0 < S; s0 += 8) {
#pragma unroll
    for (int s = 0; s < 8; ++s) {
      const size_t base = ((size_t)(b * S + s0 + s)) * 1024 + hoff + d;
      lq[s][d] = bf2f(qb[base]);
      lk[s][d] = bf2f(kb[base]);
      lv[s][d] = bf2f(vb[base]);
    }
    if (d < 8)       la[d]      = ab[((size_t)(b * S + s0 + d)) * 32 + h];
    else if (d < 16) lbt[d - 8] = ab[((size_t)(b * S + s0 + d - 8)) * 32 + 16 + h];
    __syncthreads();
#pragma unroll 1
    for (int s = 0; s < 8; ++s) {
      const float a = la[s], bet = lbt[s];
      float sk = 0.f;
#pragma unroll
      for (int e = 0; e < 64; ++e) sk += Srow[e] * lk[s][e];
      const float bd = bet * (lv[s][d] - a * sk);
      float dot = 0.f;
#pragma unroll
      for (int e = 0; e < 64; ++e) {
        Srow[e] = a * Srow[e] + bd * lk[s][e];
        dot += Srow[e] * lq[s][e];
      }
      o[((size_t)(b * S + s0 + s)) * 1024 + hoff + d] = dot;
    }
    __syncthreads();
  }
}

// --------------------------------------------------------------- group norm
__global__ __launch_bounds__(256) void k_gn_stats(const float* __restrict__ o,
                                                  float* __restrict__ stats, int S) {
  const int b = blockIdx.x >> 4;
  const int h = blockIdx.x & 15;
  const int tid = threadIdx.x;
  const int n = S * 64;
  float sm = 0.f, sq = 0.f;
  for (int i = tid; i < n; i += 256) {
    const int s = i >> 6, d = i & 63;
    const float v = o[((size_t)(b * S + s)) * 1024 + h * 64 + d];
    sm += v; sq += v * v;
  }
  __shared__ float rs[256], rq[256];
  rs[tid] = sm; rq[tid] = sq;
  __syncthreads();
  for (int st = 128; st > 0; st >>= 1) {
    if (tid < st) { rs[tid] += rs[tid + st]; rq[tid] += rq[tid + st]; }
    __syncthreads();
  }
  if (tid == 0) {
    const float inv = 1.f / (float)n;
    const float mu  = rs[0] * inv;
    const float var = rq[0] * inv - mu * mu;
    stats[blockIdx.x * 2 + 0] = mu;
    stats[blockIdx.x * 2 + 1] = rsqrtf(var + 1e-5f);
  }
}

__global__ __launch_bounds__(256) void k_gn_apply(const float* __restrict__ o,
                                                  const float* __restrict__ gate,
                                                  const float* __restrict__ stats,
                                                  const float* __restrict__ gsc,
                                                  const float* __restrict__ gbi,
                                                  unsigned short* __restrict__ ybf,
                                                  int S, int total) {
  int i = blockIdx.x * 256 + threadIdx.x;
  if (i >= total) return;
  const int c = i & 1023;
  const int row = i >> 10;
  const int b = row / S;
  const int h = c >> 6;
  const float mu  = stats[(b * 16 + h) * 2 + 0];
  const float rsg = stats[(b * 16 + h) * 2 + 1];
  const float y = (o[i] - mu) * rsg * gsc[c] + gbi[c];
  ybf[i] = f2bf(gate[i] * y);          // gate already SiLU'd by gemm epilogue
}

// ----------------------------------------------------------------- launcher
extern "C" void kernel_launch(void* const* d_in, const int* in_sizes, int n_in,
                              void* d_out, int out_size, void* d_ws, size_t ws_size,
                              hipStream_t stream) {
  (void)in_sizes; (void)n_in; (void)out_size; (void)ws_size;
  const int B = 4, S = 4096, Hd = 1024, H = 16;
  const int M = B * S;  // 16384

  const float* x   = (const float*)d_in[0];
  const float* Wq  = (const float*)d_in[1];
  const float* bq  = (const float*)d_in[2];
  const float* Wk  = (const float*)d_in[3];
  const float* bk  = (const float*)d_in[4];
  const float* Wv  = (const float*)d_in[5];
  const float* bv  = (const float*)d_in[6];
  const float* qcv = (const float*)d_in[7];
  const float* kcv = (const float*)d_in[8];
  const float* vcv = (const float*)d_in[9];
  const float* Wa  = (const float*)d_in[10];
  const float* ba  = (const float*)d_in[11];
  const float* Wb  = (const float*)d_in[12];
  const float* bb  = (const float*)d_in[13];
  const float* Wg  = (const float*)d_in[14];
  const float* bg  = (const float*)d_in[15];
  const float* gsc = (const float*)d_in[16];
  const float* gbi = (const float*)d_in[17];
  const float* Wo  = (const float*)d_in[18];
  const float* bo  = (const float*)d_in[19];
  float* out = (float*)d_out;

  char* ws = (char*)d_ws;
  size_t off = 0;
  auto alloc = [&](size_t bytes) -> char* {
    char* p = ws + off;
    off += (bytes + 255) & ~(size_t)255;
    return p;
  };
  unsigned short* xb   = (unsigned short*)alloc((size_t)M * Hd * 2);
  unsigned short* wqT  = (unsigned short*)alloc((size_t)Hd * Hd * 2);
  unsigned short* wkT  = (unsigned short*)alloc((size_t)Hd * Hd * 2);
  unsigned short* wvT  = (unsigned short*)alloc((size_t)Hd * Hd * 2);
  unsigned short* wgT  = (unsigned short*)alloc((size_t)Hd * Hd * 2);
  unsigned short* woT  = (unsigned short*)alloc((size_t)Hd * Hd * 2);
  unsigned short* wabT = (unsigned short*)alloc((size_t)32 * Hd * 2);
  float*          bab  = (float*)alloc(32 * 4);
  float*          tmp  = (float*)alloc((size_t)M * Hd * 4);   // per-projection lin
  unsigned short* qb   = (unsigned short*)alloc((size_t)M * Hd * 2);
  unsigned short* kb   = (unsigned short*)alloc((size_t)M * Hd * 2);
  unsigned short* vb   = (unsigned short*)alloc((size_t)M * Hd * 2);
  float*          gate = (float*)alloc((size_t)M * Hd * 4);
  float*          abuf = (float*)alloc((size_t)M * 32 * 4);
  float*          obuf = (float*)alloc((size_t)M * Hd * 4);
  unsigned short* ybf  = (unsigned short*)alloc((size_t)M * Hd * 2);
  float*          stats= (float*)alloc(B * H * 2 * 4);

  // --- precision prep: fp32 -> bf16, weights transposed to (N,K)
  k_f2bf<<<(M * Hd) / 256, 256, 0, stream>>>(x, xb, M * Hd);
  k_transpose_bf<<<(Hd * Hd) / 256, 256, 0, stream>>>(Wq, wqT, Hd, Hd);
  k_transpose_bf<<<(Hd * Hd) / 256, 256, 0, stream>>>(Wk, wkT, Hd, Hd);
  k_transpose_bf<<<(Hd * Hd) / 256, 256, 0, stream>>>(Wv, wvT, Hd, Hd);
  k_transpose_bf<<<(Hd * Hd) / 256, 256, 0, stream>>>(Wg, wgT, Hd, Hd);
  k_transpose_bf<<<(Hd * Hd) / 256, 256, 0, stream>>>(Wo, woT, Hd, Hd);
  k_transpose_bf<<<(Hd * H) / 256, 256, 0, stream>>>(Wa, wabT, Hd, H);
  k_transpose_bf<<<(Hd * H) / 256, 256, 0, stream>>>(Wb, wabT + (size_t)16 * Hd, Hd, H);
  k_concat_bias<<<1, 32, 0, stream>>>(ba, bb, bab);

  dim3 gG(Hd / 128, M / 128);   // (8,128) for 1024-wide GEMMs
  dim3 gAB(1, M / 128);

  // Q / K / V projections -> conv + SiLU (+ head L2 norm for q,k)
  k_gemm_bf16<<<gG, 256, 0, stream>>>(xb, wqT, bq, tmp, M, Hd, Hd, 0);
  k_conv_act<<<M, 256, 0, stream>>>(tmp, qcv, qb, S, 1);
  k_gemm_bf16<<<gG, 256, 0, stream>>>(xb, wkT, bk, tmp, M, Hd, Hd, 0);
  k_conv_act<<<M, 256, 0, stream>>>(tmp, kcv, kb, S, 1);
  k_gemm_bf16<<<gG, 256, 0, stream>>>(xb, wvT, bv, tmp, M, Hd, Hd, 0);
  k_conv_act<<<M, 256, 0, stream>>>(tmp, vcv, vb, S, 0);

  // alpha/beta (sigmoid) and out-gate (SiLU)
  k_gemm_bf16<<<gAB, 256, 0, stream>>>(xb, wabT, bab, abuf, M, 32, Hd, 1);
  k_gemm_bf16<<<gG, 256, 0, stream>>>(xb, wgT, bg, gate, M, Hd, Hd, 2);

  // sequential gated delta-rule scan, 64 independent (b,h) chains
  k_scan<<<B * H, 64, 0, stream>>>(qb, kb, vb, abuf, obuf, S);

  // group norm over (S, d) per (b,h), gate, bf16 for final GEMM
  k_gn_stats<<<B * H, 256, 0, stream>>>(obuf, stats, S);
  k_gn_apply<<<(M * Hd) / 256, 256, 0, stream>>>(obuf, gate, stats, gsc, gbi, ybf, S, M * Hd);

  // final projection -> d_out
  k_gemm_bf16<<<gG, 256, 0, stream>>>(ybf, woT, bo, out, M, Hd, Hd, 0);
}
